// GRU_84971632984744
// MI455X (gfx1250) — compile-verified, matrix operand order checked by reading, not run on hardware
//
#include <hip/hip_runtime.h>

// ---------------------------------------------------------------------------
// Persistent GRU kernel for gfx1250 (MI455X).
//   grid = 16 workgroups, 256 threads (8 wave32) each.
//   Workgroup b owns batch rows [16b, 16b+16) for all 4096 time steps.
//   W_hh pre-swizzled to bf16 WMMA B-fragments once, then held RESIDENT IN
//   VGPRS (12 frags = 96 VGPRs) for the whole kernel. h kept as fp32 in
//   registers (recurrence carry) + bf16 double-buffer in LDS (WMMA A operand).
//   One workgroup barrier per time step. Wave w computes gate col-tiles
//   {w, w+8, w+16} = r/z/n for the same 16 hidden columns -> gate math
//   entirely in registers with native v_exp/v_rcp transcendentals.
// ---------------------------------------------------------------------------

typedef __attribute__((ext_vector_type(16))) __bf16        v16bf;
typedef __attribute__((ext_vector_type(8)))  float         v8f;
typedef __attribute__((ext_vector_type(4)))  float         v4f;
typedef __attribute__((ext_vector_type(4)))  unsigned int  v4u;

union FragCast { v4u u[2]; v16bf v; };

// A-fragment: per ISA 16-bit A layout, lane needs 8 contiguous bf16 at p and
// 8 more at p+16 elements (K and K+16 halves).
static __device__ __forceinline__ v16bf lds_frag_gap(const __bf16* p) {
  FragCast t;
  t.u[0] = *(const v4u*)p;
  t.u[1] = *(const v4u*)(p + 16);
  return t.v;
}
// B-fragment: pre-swizzled so each lane's 16 bf16 are fully contiguous.
static __device__ __forceinline__ v16bf lds_frag_seq(const __bf16* p) {
  FragCast t;
  t.u[0] = *(const v4u*)p;
  t.u[1] = *(const v4u*)(p + 8);
  return t.v;
}

static __device__ __forceinline__ v8f wmma_bf16(v16bf a, v16bf b, v8f c) {
  // (neg_a, A, neg_b, B, c_mod, C, reuse_a, reuse_b)
  return __builtin_amdgcn_wmma_f32_16x16x32_bf16(false, a, false, b,
                                                 (short)0, c, false, false);
}

// sigmoid via v_exp_f32 + v_rcp_f32 (avoids IEEE div_scale/div_fmas sequence)
static __device__ __forceinline__ float fast_sigmoid(float x) {
  return __builtin_amdgcn_rcpf(1.0f + __expf(-x));
}
// tanh(x) = 2*sigmoid(2x) - 1  -> one v_exp + one v_rcp + one fma
static __device__ __forceinline__ float fast_tanh(float x) {
  return 2.0f * __builtin_amdgcn_rcpf(1.0f + __expf(-2.0f * x)) - 1.0f;
}

__global__ __launch_bounds__(256, 1)
void gru_persistent_wmma(const float* __restrict__ state,   // (32,256,128)
                         const float* __restrict__ hidden,  // (1,256,128)
                         const float* __restrict__ W_ih,    // (384,1)
                         const float* __restrict__ W_hh,    // (384,128)
                         const float* __restrict__ b_ih,    // (384)
                         const float* __restrict__ b_hh,    // (384)
                         const float* __restrict__ W_lin,   // (64,128)
                         const float* __restrict__ b_lin,   // (64)
                         float* __restrict__ out)           // outs | hT
{
  constexpr int H_ = 128, B_ = 256, T_ = 32, L_ = 128, O_ = 64;

  extern __shared__ char smem[];
  __bf16* whhF  = (__bf16*)smem;              //  96 frags * 512 bf16 = 96 KB
  __bf16* wlinF = whhF  + 96 * 512;           //  16 frags * 512 bf16 = 16 KB
  __bf16* hbuf0 = wlinF + 16 * 512;           //  16*128 bf16 = 4 KB
  __bf16* hbuf1 = hbuf0 + 16 * 128;           //  16*128 bf16 = 4 KB
  float*  xbuf  = (float*)(hbuf1 + 16 * 128); //  128*16 f32  = 8 KB
                                              //  total 128 KB dyn LDS

  const int tid  = threadIdx.x;
  const int w    = tid >> 5;          // wave 0..7
  const int lane = tid & 31;
  const int l16  = lane & 15;
  const int hi   = lane >> 4;         // half-wave
  const int wgBase = blockIdx.x * 16; // batch-row base
  const int j    = w * 16 + l16;      // hidden column owned by this lane

  // ---- pack W_hh into bf16 WMMA B-fragments (B[k][n] = W_hh[n][k]) --------
  // frag f = nt*4 + kb ; per-lane 16 values contiguous at f*512 + ll*16.
  for (int e = tid; e < 96 * 512; e += 256) {
    int f  = e >> 9, rr = e & 511;
    int ll = rr >> 4, v = rr & 15;
    int nt = f >> 2,  kb = f & 3;
    int n  = nt * 16 + (ll & 15);
    int h8 = (ll >> 4) * 8;
    int k  = kb * 32 + ((v < 8) ? (h8 + v) : (16 + h8 + (v - 8)));
    whhF[e] = (__bf16)W_hh[n * H_ + k];
  }
  // ---- pack W_lin the same way (B[k][o] = W_lin[o][k]) --------------------
  for (int e = tid; e < 16 * 512; e += 256) {
    int f  = e >> 9, rr = e & 511;
    int ll = rr >> 4, v = rr & 15;
    int ot = f >> 2,  kb = f & 3;
    int o  = ot * 16 + (ll & 15);
    int h8 = (ll >> 4) * 8;
    int k  = kb * 32 + ((v < 8) ? (h8 + v) : (16 + h8 + (v - 8)));
    wlinF[e] = (__bf16)W_lin[o * H_ + k];
  }

  // ---- per-lane gate constants (depend only on column j) ------------------
  const float wi_r = W_ih[j], wi_z = W_ih[j + H_], wi_n = W_ih[j + 2 * H_];
  const float bi_r = b_ih[j], bi_z = b_ih[j + H_], bi_n = b_ih[j + 2 * H_];
  const float bh_r = b_hh[j], bh_z = b_hh[j + H_], bh_n = b_hh[j + 2 * H_];
  const float blin = b_lin[(w & 3) * 16 + l16];

  // ---- init h: fp32 registers (C-layout) + bf16 LDS copy ------------------
  float hreg[8];
#pragma unroll
  for (int r = 0; r < 8; ++r) {
    int m = r + 8 * hi;
    hreg[r] = hidden[(size_t)(wgBase + m) * H_ + j];
    hbuf0[m * H_ + j] = (__bf16)hreg[r];
  }
  __syncthreads();

  // ---- hoist W_hh B-fragments into registers: loop-invariant for 4096 steps
  const int ntR = w, ntZ = w + 8, ntN = w + 16;
  v16bf bR[4], bZ[4], bN[4];
#pragma unroll
  for (int kb = 0; kb < 4; ++kb) {
    bR[kb] = lds_frag_seq(whhF + (ntR * 4 + kb) * 512 + lane * 16);
    bZ[kb] = lds_frag_seq(whhF + (ntZ * 4 + kb) * 512 + lane * 16);
    bN[kb] = lds_frag_seq(whhF + (ntN * 4 + kb) * 512 + lane * 16);
  }

  for (int t = 0; t < T_; ++t) {
    // stage x block transposed: xbuf[l][m] = state[t][wgBase+m][l]
    {
      int m  = tid >> 4;
      int l0 = (tid & 15) * 8;
      const float* sp = state + (((size_t)t * B_) + wgBase + m) * L_ + l0;
#pragma unroll
      for (int i = 0; i < 8; ++i) xbuf[(l0 + i) * 16 + m] = sp[i];
    }
    __syncthreads();

#pragma unroll 2
    for (int l = 0; l < L_; ++l) {
      const __bf16* hsrc = (l & 1) ? hbuf1 : hbuf0;
      __bf16*       hdst = (l & 1) ? hbuf0 : hbuf1;

      // per-row input scalars for this lane's 8 rows
      v4f x0 = *(const v4f*)(xbuf + l * 16 + hi * 8);
      v4f x1 = *(const v4f*)(xbuf + l * 16 + hi * 8 + 4);

      // gh tiles: r/z/n for the same 16 hidden columns, K=128 in 4 WMMAs each
      v8f accR = {}, accZ = {}, accN = {};
#pragma unroll
      for (int kb = 0; kb < 4; ++kb) {
        v16bf a = lds_frag_gap(hsrc + l16 * H_ + kb * 32 + hi * 8);
        accR = wmma_bf16(a, bR[kb], accR);
        accZ = wmma_bf16(a, bZ[kb], accZ);
        accN = wmma_bf16(a, bN[kb], accN);
      }

      // gates entirely in registers; fp32 recurrence carry in hreg
#pragma unroll
      for (int r = 0; r < 8; ++r) {
        float x  = (r < 4) ? x0[r & 3] : x1[r & 3];
        float rg = fast_sigmoid((x * wi_r + bi_r) + (accR[r] + bh_r));
        float zg = fast_sigmoid((x * wi_z + bi_z) + (accZ[r] + bh_z));
        float ng = fast_tanh((x * wi_n + bi_n) + rg * (accN[r] + bh_n));
        float hn = ng + zg * (hreg[r] - ng);        // (1-z)*n + z*h
        hreg[r] = hn;
        hdst[(r + 8 * hi) * H_ + j] = (__bf16)hn;   // next step's A operand
      }
      __syncthreads();   // single barrier per step (h double-buffered)
    }

    // ---- outs[t] = h @ W_lin.T + b_lin  (h is in hbuf0: 128 steps = even) --
    if (w < 4) {   // wave-uniform branch: EXEC all-ones inside (WMMA-legal)
      v8f acc = {};
#pragma unroll
      for (int kb = 0; kb < 4; ++kb) {
        v16bf a = lds_frag_gap(hbuf0 + l16 * H_ + kb * 32 + hi * 8);
        acc = wmma_bf16(a, lds_frag_seq(wlinF + (w * 4 + kb) * 512 + lane * 16), acc);
      }
      float* op = out + ((size_t)t * B_ + wgBase) * O_;
#pragma unroll
      for (int r = 0; r < 8; ++r)
        op[(r + 8 * hi) * O_ + (w * 16 + l16)] = acc[r] + blin;
    }
  }

  // ---- final hidden state hT (fp32 from registers) ------------------------
  float* hT = out + (size_t)T_ * B_ * O_;
#pragma unroll
  for (int r = 0; r < 8; ++r)
    hT[(size_t)(wgBase + r + 8 * hi) * H_ + j] = hreg[r];
}

extern "C" void kernel_launch(void* const* d_in, const int* in_sizes, int n_in,
                              void* d_out, int out_size, void* d_ws, size_t ws_size,
                              hipStream_t stream) {
  const float* state  = (const float*)d_in[0];
  const float* hidden = (const float*)d_in[1];
  const float* W_ih   = (const float*)d_in[2];
  const float* W_hh   = (const float*)d_in[3];
  const float* b_ih   = (const float*)d_in[4];
  const float* b_hh   = (const float*)d_in[5];
  const float* W_lin  = (const float*)d_in[6];
  const float* b_lin  = (const float*)d_in[7];
  float* out = (float*)d_out;

  // 96*512*2 + 16*512*2 + 2*(16*128*2) + 128*16*4 = 131072 bytes of LDS
  constexpr size_t SMEM = 131072;
  gru_persistent_wmma<<<16, 256, SMEM, stream>>>(
      state, hidden, W_ih, W_hh, b_ih, b_hh, W_lin, b_lin, out);
}
